// Net_41412074668367
// MI455X (gfx1250) — matrix operand order repaired and checked by reading, not compiled, not run on hardware
//
#include <hip/hip_runtime.h>
#include <hip/hip_bf16.h>

typedef __attribute__((ext_vector_type(16))) _Float16 v16h;
typedef __attribute__((ext_vector_type(8)))  float    v8f;

#define H_F   37
#define W_F   12
#define C_F   25
#define NCCK  2048   // padded K for stage-2: [c(25)][d(5)][k pad 16] = 2000 -> 2048
#define KCH   64     // 2048/32

// LDS layout (bytes) for k_ncc_red1
#define OFF_F1S  0        // [25][5][12] f32  : 6000  (pad 6016)
#define OFF_F2S  6016     // [25][9][12] f32  : 10800 (pad 10816)
#define OFF_XN   16832    // [25][32][16] f16 : 25600
#define OFF_YN   42432    // [5][25][16][32] f16 : 128000
#define OFF_NCC  170432   // [2048][16] f16   : 65536
#define OFF_PART 235968   // [8][256] f32     : 8192
#define LDS_TOT  244160

static __device__ __forceinline__ v8f wmma_f16(v16h a, v16h b, v8f c) {
    return __builtin_amdgcn_wmma_f32_16x16x32_f16(false, a, false, b, (short)0, c, false, false);
}

// ---------------------------------------------------------------------------
// K0: repack w_red1 [25,1500] -> f16 [32][2048] with per-(c,d) K padded 12->16
// ---------------------------------------------------------------------------
__global__ void k_wred1_cvt(const float* __restrict__ w, _Float16* __restrict__ wf) {
    int t = blockIdx.x * blockDim.x + threadIdx.x;
    if (t >= 32 * NCCK) return;
    int o = t / NCCK, kk = t % NCCK;
    int c = kk / 80, d = (kk / 16) % 5, k = kk % 16;
    float v = (o < 25 && c < 25 && k < 12) ? w[o * 1500 + c * 60 + d * 12 + k] : 0.0f;
    wf[t] = (_Float16)v;
}

// ---------------------------------------------------------------------------
// K1: conv1(5x5 VALID, 3->20) + relu + maxpool2 for both branches (bb: 0..127)
// ---------------------------------------------------------------------------
__global__ void k_conv1pool(const float* __restrict__ x1, const float* __restrict__ x2,
                            const float* __restrict__ w, const float* __restrict__ bias,
                            float* __restrict__ out) {
    int t = blockIdx.x * blockDim.x + threadIdx.x;
    const int N = 128 * 20 * 78 * 28;
    if (t >= N) return;
    int j  = t % 28;
    int i  = (t / 28) % 78;
    int oc = (t / (28 * 78)) % 20;
    int bb = t / (28 * 78 * 20);
    const float* x = (bb < 64) ? (x1 + (size_t)bb * 3 * 160 * 60)
                               : (x2 + (size_t)(bb - 64) * 3 * 160 * 60);
    float m = -1e30f;
    for (int py = 0; py < 2; ++py)
    for (int px = 0; px < 2; ++px) {
        int oy = 2 * i + py, ox = 2 * j + px;
        float acc = bias[oc];
        for (int ic = 0; ic < 3; ++ic)
        for (int ky = 0; ky < 5; ++ky)
        for (int kx = 0; kx < 5; ++kx)
            acc += x[((size_t)ic * 160 + oy + ky) * 60 + ox + kx] *
                   w[((oc * 3 + ic) * 5 + ky) * 5 + kx];
        acc = acc > 0.0f ? acc : 0.0f;
        m = acc > m ? acc : m;
    }
    out[t] = m;
}

// ---------------------------------------------------------------------------
// K2: conv2(5x5 VALID, 20->25) + relu + maxpool2
// ---------------------------------------------------------------------------
__global__ void k_conv2pool(const float* __restrict__ in, const float* __restrict__ w,
                            const float* __restrict__ bias, float* __restrict__ out) {
    int t = blockIdx.x * blockDim.x + threadIdx.x;
    const int N = 128 * 25 * 37 * 12;
    if (t >= N) return;
    int j  = t % 12;
    int i  = (t / 12) % 37;
    int oc = (t / 444) % 25;
    int bb = t / (444 * 25);
    const float* xin = in + (size_t)bb * 20 * 78 * 28;
    float m = -1e30f;
    for (int py = 0; py < 2; ++py)
    for (int px = 0; px < 2; ++px) {
        int oy = 2 * i + py, ox = 2 * j + px;
        float acc = bias[oc];
        for (int ic = 0; ic < 20; ++ic)
        for (int ky = 0; ky < 5; ++ky)
        for (int kx = 0; kx < 5; ++kx)
            acc += xin[((size_t)ic * 78 + oy + ky) * 28 + ox + kx] *
                   w[((oc * 20 + ic) * 5 + ky) * 5 + kx];
        acc = acc > 0.0f ? acc : 0.0f;
        m = acc > m ? acc : m;
    }
    out[t] = m;
}

// ---------------------------------------------------------------------------
// K3: fused normxcorr + relu + 1x1 reduction conv (1500->25) + relu.
// One block per (b, row). Padded layouts -> branch-free WMMA operand loads
// and branch-free D stores (pad lanes carry exact zeros).
// ---------------------------------------------------------------------------
__global__ void k_ncc_red1(const float* __restrict__ f1, const float* __restrict__ f2,
                           const _Float16* __restrict__ wf, const float* __restrict__ bias,
                           float* __restrict__ out) {
    extern __shared__ char smem[];
    float*    f1s = (float*)(smem + OFF_F1S);
    float*    f2s = (float*)(smem + OFF_F2S);
    _Float16* Xn  = (_Float16*)(smem + OFF_XN);
    _Float16* Yn  = (_Float16*)(smem + OFF_YN);
    _Float16* NCC = (_Float16*)(smem + OFF_NCC);
    float*    part= (float*)(smem + OFF_PART);

    const int i0 = blockIdx.x;      // 0..36
    const int b  = blockIdx.y;      // 0..63
    const int tid  = threadIdx.x;
    const int lane = tid & 31;
    const int wv   = tid >> 5;      // 0..7 (wave32)
    const int g    = lane >> 4;

    const float* F1 = f1 + (size_t)b * C_F * H_F * W_F;
    const float* F2 = f2 + (size_t)(b + 64) * C_F * H_F * W_F;

    // ---- phase 0: zero Xn+Yn (contiguous span) + NCC tail rows, load slabs ----
    {
        uint4 z; z.x = z.y = z.z = z.w = 0u;
        uint4* span = (uint4*)(smem + OFF_XN);          // Xn..Yn = 153600 B
        for (int t = tid; t < (25600 + 128000) / 16; t += 256) span[t] = z;
        uint4* tail = (uint4*)(smem + OFF_NCC + 2000 * 16 * 2);  // rows 2000..2047
        for (int t = tid; t < (48 * 16 * 2) / 16; t += 256) tail[t] = z;
    }
    for (int t = tid; t < 1500; t += 256) {             // f1 rows i0-2..i0+2
        int c = t / 60, u = (t / 12) % 5, j = t % 12;
        int row = i0 - 2 + u;
        f1s[t] = (row >= 0 && row < H_F) ? F1[((size_t)c * H_F + row) * W_F + j] : 0.0f;
    }
    for (int t = tid; t < 2700; t += 256) {             // f2 rows i0-4..i0+4
        int c = t / 108, s = (t / 12) % 9, j = t % 12;
        int row = i0 - 4 + s;
        f2s[t] = (row >= 0 && row < H_F) ? F2[((size_t)c * H_F + row) * W_F + j] : 0.0f;
    }
    __syncthreads();

    // ---- phase 1: zero-mean/unit-std 5x5 patches (population std, +0.01) ----
    for (int t = tid; t < 1800; t += 256) {
        float vals[25];
        if (t < 300) {                                  // Xn task (c, j)
            int c = t / 12, j = t % 12;
            float s = 0.0f, s2 = 0.0f;
            for (int u = 0; u < 5; ++u)
            for (int v = 0; v < 5; ++v) {
                int col = j + v - 2;
                float x = (col >= 0 && col < 12) ? f1s[c * 60 + u * 12 + col] : 0.0f;
                vals[u * 5 + v] = x; s += x; s2 += x * x;
            }
            float mu = s * 0.04f;
            float var = s2 * 0.04f - mu * mu; var = var > 0.0f ? var : 0.0f;
            float inv = 1.0f / (__builtin_sqrtf(var) + 0.01f);
            for (int p = 0; p < 25; ++p)
                Xn[(c * 32 + p) * 16 + j] = (_Float16)((vals[p] - mu) * inv);
        } else {                                        // Yn task (d, c, k)
            int t2 = t - 300;
            int d = t2 / 300, c = (t2 / 12) % 25, k = t2 % 12;
            int yr = i0 + d - 2;
            if (yr >= 0 && yr < H_F) {                  // else stays zero (Ynp row pad)
                float s = 0.0f, s2 = 0.0f;
                for (int u = 0; u < 5; ++u)
                for (int v = 0; v < 5; ++v) {
                    int col = k + v - 2;
                    float x = (col >= 0 && col < 12) ? f2s[c * 108 + (d + u) * 12 + col] : 0.0f;
                    vals[u * 5 + v] = x; s += x; s2 += x * x;
                }
                float mu = s * 0.04f;
                float var = s2 * 0.04f - mu * mu; var = var > 0.0f ? var : 0.0f;
                float inv = 1.0f / (__builtin_sqrtf(var) + 0.01f);
                for (int p = 0; p < 25; ++p)
                    Yn[((d * 25 + c) * 16 + k) * 32 + p] = (_Float16)((vals[p] - mu) * inv);
            }
        }
    }
    __syncthreads();

    // ---- stage 1: ncc[c,d,k,j] = (1/25) * Xn[c,j,:]·Yn[d,c,k,:], relu -> LDS ----
    // A = Yn (M=k pad16, K=p pad32), B = Xn (K=p, N=j pad16): all pads are zeros,
    // so loads AND stores are branch-free.
    for (int idx = wv; idx < 125; idx += 8) {
        int c = idx / 5, dd = idx % 5;
        v16h a, bb;
        {
            int k = lane & 15;
            int base = ((dd * 25 + c) * 16 + k) * 32;
            for (int h = 0; h < 8; ++h) {
                a[h]     = Yn[base + 8 * g + h];
                a[h + 8] = Yn[base + 16 + 8 * g + h];
            }
        }
        {
            int base = (c * 32) * 16 + (lane & 15);     // B col j = lane&15, elements = p (table order)
            for (int n = 0; n < 16; ++n) bb[n] = Xn[base + ((n < 8) ? (8 * g + n) : (16 + 8 * g + (n - 8))) * 16];
        }
        v8f cz; for (int r = 0; r < 8; ++r) cz[r] = 0.0f;
        v8f dacc = wmma_f16(a, bb, cz);
        int sbase = ((c * 80 + dd * 16 + 8 * g) * 16) + (lane & 15);
        for (int r = 0; r < 8; ++r) {                   // rows kk = r + 8g (pad rows = 0)
            float v = dacc[r] * 0.04f;                  // /25
            NCC[sbase + r * 16] = (_Float16)(v > 0.0f ? v : 0.0f);
        }
    }
    __syncthreads();

    // ---- stage 2: out[o,j] = sum_K w[o,K] * NCC[K,j]  (K = 2048 padded) ----
    {
        int ot = wv & 1;                // o tile: 0 -> o 0..15, 1 -> o 16..31
        int cw = wv >> 1;               // chunk class 0..3
        v8f acc; for (int r = 0; r < 8; ++r) acc[r] = 0.0f;
        for (int chunk = cw; chunk < KCH; chunk += 4) {
            int kbase = chunk * 32;
            v16h a, bb;
            int o = ot * 16 + (lane & 15);              // rows 25..31 zero-padded
            const _Float16* wrow = wf + (size_t)o * NCCK + kbase;
            for (int h = 0; h < 8; ++h) {
                a[h]     = wrow[8 * g + h];
                a[h + 8] = wrow[16 + 8 * g + h];
            }
            int base = kbase * 16 + (lane & 15);        // B col = j (lane&15), halfs = K (table order)
            for (int n = 0; n < 16; ++n) bb[n] = NCC[base + ((n < 8) ? (8 * g + n) : (16 + 8 * g + (n - 8))) * 16];
            acc = wmma_f16(a, bb, acc);
        }
        for (int r = 0; r < 8; ++r) part[wv * 256 + r * 32 + lane] = acc[r];
    }
    __syncthreads();

    // ---- reduce 4 partials per o-tile, add bias, relu, store ----
    for (int e = tid; e < 512; e += 256) {
        int ot = e >> 8, idx = e & 255;
        int r = idx >> 5, l2 = idx & 31;
        float v = part[(ot + 0) * 256 + idx] + part[(ot + 2) * 256 + idx] +
                  part[(ot + 4) * 256 + idx] + part[(ot + 6) * 256 + idx];
        int o = ot * 16 + r + 8 * (l2 >> 4);
        int j = l2 & 15;
        if (o < 25 && j < 12) {
            float y = v + bias[o];
            out[(((size_t)b * 25 + o) * H_F + i0) * W_F + j] = y > 0.0f ? y : 0.0f;
        }
    }
}

// ---------------------------------------------------------------------------
// K4: conv 3x3 VALID (25->25) + maxpool2 (no relu) -> red2p[64,2125]
// ---------------------------------------------------------------------------
__global__ void k_red2pool(const float* __restrict__ in, const float* __restrict__ w,
                           const float* __restrict__ bias, float* __restrict__ out) {
    int t = blockIdx.x * blockDim.x + threadIdx.x;
    const int N = 64 * 25 * 17 * 5;
    if (t >= N) return;
    int j  = t % 5;
    int i  = (t / 5) % 17;
    int oc = (t / 85) % 25;
    int b  = t / (85 * 25);
    const float* xin = in + (size_t)b * 25 * 37 * 12;
    float m = -1e30f;
    for (int py = 0; py < 2; ++py)
    for (int px = 0; px < 2; ++px) {
        int oy = 2 * i + py, ox = 2 * j + px;
        float acc = bias[oc];
        for (int ic = 0; ic < 25; ++ic)
        for (int ky = 0; ky < 3; ++ky)
        for (int kx = 0; kx < 3; ++kx)
            acc += xin[((size_t)ic * 37 + oy + ky) * 12 + ox + kx] *
                   w[((oc * 25 + ic) * 3 + ky) * 3 + kx];
        m = acc > m ? acc : m;
    }
    out[(size_t)b * 2125 + oc * 85 + i * 5 + j] = m;
}

// ---------------------------------------------------------------------------
// K5: linear1 via WMMA: out[64,500] = act[64,2125] @ w[500,2125]^T + b
// ---------------------------------------------------------------------------
__global__ void k_lin1(const float* __restrict__ act, const float* __restrict__ w,
                       const float* __restrict__ bias, float* __restrict__ out) {
    const int nt = blockIdx.x, mt = blockIdx.y;
    const int lane = threadIdx.x & 31;
    const int g = lane >> 4;
    const int K = 2125;
    const bool full_n = (nt < 31);
    v8f acc; for (int r = 0; r < 8; ++r) acc[r] = 0.0f;
    const int m = mt * 16 + (lane & 15);
    // full chunks: k < 2112 (no predicates)
    for (int chunk = 0; chunk < 66; ++chunk) {
        int kbase = chunk * 32;
        v16h a, bb;
        const float* arow = act + (size_t)m * K + kbase;
        for (int h = 0; h < 8; ++h) {
            a[h]     = (_Float16)arow[8 * g + h];
            a[h + 8] = (_Float16)arow[16 + 8 * g + h];
        }
        int kb = kbase;
        if (full_n) {
            for (int n = 0; n < 16; ++n)
                bb[n] = (_Float16)w[(size_t)(nt * 16 + (lane & 15)) * K + kb + ((n < 8) ? (8 * g + n) : (16 + 8 * g + (n - 8)))];
        } else {
            for (int n = 0; n < 16; ++n) {
                int nn = nt * 16 + (lane & 15);
                bb[n] = (nn < 500) ? (_Float16)w[(size_t)nn * K + kb + ((n < 8) ? (8 * g + n) : (16 + 8 * g + (n - 8)))] : (_Float16)0.0f;
            }
        }
        acc = wmma_f16(a, bb, acc);
    }
    // tail chunk: kbase = 2112, valid k < 2125
    {
        int kbase = 66 * 32;
        v16h a, bb;
        for (int h = 0; h < 16; ++h) {
            int k = kbase + ((h < 8) ? (8 * g + h) : (16 + 8 * g + (h - 8)));
            a[h] = (k < K) ? (_Float16)act[(size_t)m * K + k] : (_Float16)0.0f;
        }
        for (int n = 0; n < 16; ++n) {
            int kb = kbase + ((n < 8) ? (8 * g + n) : (16 + 8 * g + (n - 8)));
            int nn = nt * 16 + (lane & 15);
            bb[n] = (kb < K && nn < 500) ? (_Float16)w[(size_t)nn * K + kb] : (_Float16)0.0f;
        }
        acc = wmma_f16(a, bb, acc);
    }
    for (int r = 0; r < 8; ++r) {
        int mm = mt * 16 + r + 8 * g;
        int n = nt * 16 + (lane & 15);
        if (n < 500) out[(size_t)mm * 500 + n] = acc[r] + bias[n];
    }
}

// ---------------------------------------------------------------------------
// K6: linear2: out[64,2] = lin1out[64,500] @ w[2,500]^T + b
// ---------------------------------------------------------------------------
__global__ void k_lin2(const float* __restrict__ in, const float* __restrict__ w,
                       const float* __restrict__ bias, float* __restrict__ out) {
    int t = threadIdx.x;
    if (t >= 128) return;
    int m = t >> 1, n = t & 1;
    float acc = bias[n];
    for (int k = 0; k < 500; ++k) acc += in[(size_t)m * 500 + k] * w[n * 500 + k];
    out[m * 2 + n] = acc;
}

// ---------------------------------------------------------------------------
extern "C" void kernel_launch(void* const* d_in, const int* in_sizes, int n_in,
                              void* d_out, int out_size, void* d_ws, size_t ws_size,
                              hipStream_t stream) {
    (void)in_sizes; (void)n_in; (void)out_size; (void)ws_size;
    const float* x1      = (const float*)d_in[0];
    const float* x2      = (const float*)d_in[1];
    const float* w_conv1 = (const float*)d_in[2];
    const float* b_conv1 = (const float*)d_in[3];
    const float* w_conv2 = (const float*)d_in[4];
    const float* b_conv2 = (const float*)d_in[5];
    const float* w_red1  = (const float*)d_in[6];
    const float* b_red1  = (const float*)d_in[7];
    const float* w_red2  = (const float*)d_in[8];
    const float* b_red2  = (const float*)d_in[9];
    const float* w_lin1  = (const float*)d_in[10];
    const float* b_lin1  = (const float*)d_in[11];
    const float* w_lin2  = (const float*)d_in[12];
    const float* b_lin2  = (const float*)d_in[13];
    float* out = (float*)d_out;

    char* ws = (char*)d_ws;
    size_t off = 0;
    auto carve = [&](size_t bytes) { char* p = ws + off; off = (off + bytes + 255) & ~(size_t)255; return p; };
    _Float16* wf16   = (_Float16*)carve((size_t)32 * NCCK * 2);       // 128 KB
    float* pool1     = (float*)carve((size_t)128 * 20 * 78 * 28 * 4); // 22.4 MB
    float* f_both    = (float*)carve((size_t)128 * 25 * 37 * 12 * 4); //  5.7 MB
    float* red1out   = (float*)carve((size_t)64 * 25 * 37 * 12 * 4);  //  2.8 MB
    float* red2p     = (float*)carve((size_t)64 * 2125 * 4);          //  0.5 MB
    float* lin1out   = (float*)carve((size_t)64 * 500 * 4);           //  0.1 MB

    k_wred1_cvt<<<(32 * NCCK + 255) / 256, 256, 0, stream>>>(w_red1, wf16);
    {
        int N = 128 * 20 * 78 * 28;
        k_conv1pool<<<(N + 255) / 256, 256, 0, stream>>>(x1, x2, w_conv1, b_conv1, pool1);
    }
    {
        int N = 128 * 25 * 37 * 12;
        k_conv2pool<<<(N + 255) / 256, 256, 0, stream>>>(pool1, w_conv2, b_conv2, f_both);
    }
    {
        hipFuncSetAttribute(reinterpret_cast<const void*>(k_ncc_red1),
                            hipFuncAttributeMaxDynamicSharedMemorySize, (int)LDS_TOT);
        k_ncc_red1<<<dim3(37, 64), 256, LDS_TOT, stream>>>(f_both, f_both, wf16, b_red1, red1out);
    }
    {
        int N = 64 * 25 * 17 * 5;
        k_red2pool<<<(N + 255) / 256, 256, 0, stream>>>(red1out, w_red2, b_red2, red2p);
    }
    k_lin1<<<dim3(32, 4), 32, 0, stream>>>(red2p, w_lin1, b_lin1, lin1out);
    k_lin2<<<1, 128, 0, stream>>>(lin1out, w_lin2, b_lin2, out);
}
